// CrossAttentionBlock_52140902973755
// MI455X (gfx1250) — matrix operand order repaired and checked by reading, not compiled, not run on hardware
//
#include <hip/hip_runtime.h>
#include <hip/hip_bf16.h>

typedef __attribute__((ext_vector_type(16))) __bf16 v16bf;
typedef __attribute__((ext_vector_type(8)))  __bf16 bf16x8;
typedef __attribute__((ext_vector_type(8)))  float  v8f;

#define B_    2
#define C_    512
#define NQ_   4096
#define CTX_  1024
#define CD_   768
#define H_    8
#define HD_   64
#define G_    32
#define GEL_  (16 * NQ_)   /* elements per (batch, group) = 65536 */
#define EPS_  1e-5f
#define SCL_  0.125f       /* HEAD_DIM^-0.5 */

static __device__ __forceinline__ v8f zero8() {
  v8f z;
#pragma unroll
  for (int i = 0; i < 8; ++i) z[i] = 0.f;
  return z;
}

static __device__ __forceinline__ v8f wmma_bf16(v16bf a, v16bf b, v8f c) {
  // D(f32,16x16) = A(bf16,16x32) * B(bf16,32x16) + C
  return __builtin_amdgcn_wmma_f32_16x16x32_bf16(false, a, false, b, (short)0, c,
                                                 false, false);
}

// A fragment (16x32 bf16, row-major source, leading dim = lda elements).
// Lane L: M = L%16, kb = (L>=16 ? 8 : 0); elems 0..7 = A[M][kb..kb+7],
// elems 8..15 = A[M][kb+16..kb+23]  (ISA 7.12.2 16-bit A layout).
static __device__ __forceinline__ v16bf load_a_frag(const __bf16* __restrict__ A, int lda) {
  int lane = threadIdx.x & 31;
  int m = lane & 15;
  int kb = (lane >= 16) ? 8 : 0;
  const __bf16* p = A + (size_t)m * lda + kb;
  bf16x8 lo = *(const bf16x8*)(p);
  bf16x8 hi = *(const bf16x8*)(p + 16);
  v16bf a;
#pragma unroll
  for (int i = 0; i < 8; ++i) { a[i] = lo[i]; a[i + 8] = hi[i]; }
  return a;
}

// B fragment from transposed operand Bt stored [N x K] row-major.
// Lane L: N = L%16, kb = (L>=16 ? 8 : 0); elems 0..7 -> K = kb..kb+7, elems 8..15 -> K = kb+16..kb+23.
static __device__ __forceinline__ v16bf load_b_frag(const __bf16* __restrict__ Bt, int ldb) {
  int lane = threadIdx.x & 31;
  int n = lane & 15;
  int kb = (lane >= 16) ? 8 : 0;
  const __bf16* p = Bt + (size_t)n * ldb + kb;
  bf16x8 lo = *(const bf16x8*)(p);
  bf16x8 hi = *(const bf16x8*)(p + 16);
  v16bf b;
#pragma unroll
  for (int i = 0; i < 8; ++i) { b[i] = lo[i]; b[i + 8] = hi[i]; }
  return b;
}

// ---------------- GroupNorm ----------------

__global__ void gn_stats_kernel(const float* __restrict__ x, float* __restrict__ stats) {
  int bg = blockIdx.x;  // b*32 + g ; each group is a contiguous 65536-float block
  const float* p = x + (size_t)bg * GEL_;
  float s = 0.f, ss = 0.f;
  for (int i = threadIdx.x; i < GEL_; i += blockDim.x) {
    float v = p[i];
    s += v; ss += v * v;
  }
  __shared__ float sb[256], ssb[256];
  sb[threadIdx.x] = s; ssb[threadIdx.x] = ss;
  __syncthreads();
  for (int off = 128; off > 0; off >>= 1) {
    if (threadIdx.x < off) {
      sb[threadIdx.x] += sb[threadIdx.x + off];
      ssb[threadIdx.x] += ssb[threadIdx.x + off];
    }
    __syncthreads();
  }
  if (threadIdx.x == 0) {
    float mean = sb[0] * (1.f / GEL_);
    float var = ssb[0] * (1.f / GEL_) - mean * mean;
    stats[bg * 2 + 0] = mean;
    stats[bg * 2 + 1] = rsqrtf(var + EPS_);
  }
}

// normalize + transpose [b,c,n] -> bf16 [b,n,c]
__global__ void gn_apply_kernel(const float* __restrict__ x,
                                const float* __restrict__ stats,
                                const float* __restrict__ gamma,
                                const float* __restrict__ beta,
                                __bf16* __restrict__ xnT) {
  size_t idx = (size_t)blockIdx.x * blockDim.x + threadIdx.x;  // = (b*NQ_+n)*C_ + c
  if (idx >= (size_t)B_ * NQ_ * C_) return;
  int c = (int)(idx % C_);
  int n = (int)((idx / C_) % NQ_);
  int b = (int)(idx / ((size_t)C_ * NQ_));
  float xv = x[((size_t)b * C_ + c) * NQ_ + n];
  int g = c >> 4;
  float mu = stats[(b * G_ + g) * 2 + 0];
  float rs = stats[(b * G_ + g) * 2 + 1];
  xnT[idx] = (__bf16)((xv - mu) * rs * gamma[c] + beta[c]);
}

// ---------------- format conversion ----------------

__global__ void convert_bf16_kernel(const float* __restrict__ src, __bf16* __restrict__ dst,
                                    int count) {
  int idx = blockIdx.x * blockDim.x + threadIdx.x;
  if (idx < count) dst[idx] = (__bf16)src[idx];
}

// [K x N] f32 -> [N x K] bf16 (dst write-coalesced)
__global__ void transpose_bf16_kernel(const float* __restrict__ src, __bf16* __restrict__ dst,
                                      int K, int N) {
  int idx = blockIdx.x * blockDim.x + threadIdx.x;
  if (idx >= K * N) return;
  int k = idx % K;
  int n = idx / K;
  dst[idx] = (__bf16)src[(size_t)k * N + n];  // dst[n*K+k]
}

// ---------------- generic WMMA GEMM ----------------
// One wave computes a 32(M) x 64(N) tile of C = A[MxK] * Bt[NxK]^T per batch:
// 2 A-fragments share 4 B-fragments per K-step -> 8 WMMAs per 12 b128 loads.
// mode 0: Q  bf16 [b][h][m][d]        mode 1: K bf16 [b][h][m][d]
// mode 2: Vt bf16 [b][h][d][m]        mode 3: f32 out[b][n][m] + bo[n] + residual
__global__ void __launch_bounds__(128)
gemm_wmma_kernel(const __bf16* __restrict__ A, const __bf16* __restrict__ Bt,
                 int M, int K, int strideA, int mode,
                 void* __restrict__ outp, const float* __restrict__ bo,
                 const float* __restrict__ xres) {
  const int nt = C_ / 64;  // N = 512 for every GEMM here
  int wave = threadIdx.x >> 5;
  int lane = threadIdx.x & 31;
  int t = blockIdx.x * 4 + wave;
  int mt = M / 32;
  int nT = t % nt;
  int mT = (t / nt) % mt;
  int b = t / (nt * mt);

  const __bf16* Ab = A + (size_t)b * strideA + (size_t)mT * 32 * K;
  v8f acc[2][4];
#pragma unroll
  for (int mi = 0; mi < 2; ++mi)
#pragma unroll
    for (int i = 0; i < 4; ++i) acc[mi][i] = zero8();

  for (int k0 = 0; k0 < K; k0 += 32) {
    // stream the next A tiles through L2 while we compute (global_prefetch_b8)
    __builtin_prefetch(Ab + k0 + 128, 0, 1);
    __builtin_prefetch(Ab + (size_t)16 * K + k0 + 128, 0, 1);
    v16bf a0 = load_a_frag(Ab + k0, K);
    v16bf a1 = load_a_frag(Ab + (size_t)16 * K + k0, K);
#pragma unroll
    for (int t4 = 0; t4 < 4; ++t4) {
      v16bf bf = load_b_frag(Bt + (size_t)(nT * 64 + t4 * 16) * K + k0, K);
      acc[0][t4] = wmma_bf16(a0, bf, acc[0][t4]);
      acc[1][t4] = wmma_bf16(a1, bf, acc[1][t4]);
    }
  }

  int col = lane & 15;
  int rowoff = (lane >= 16) ? 8 : 0;
#pragma unroll
  for (int mi = 0; mi < 2; ++mi) {
#pragma unroll
    for (int t4 = 0; t4 < 4; ++t4) {
      int n = nT * 64 + t4 * 16 + col;
#pragma unroll
      for (int r = 0; r < 8; ++r) {
        int m = mT * 32 + mi * 16 + r + rowoff;
        float v = acc[mi][t4][r];
        if (mode == 0) {
          ((__bf16*)outp)[(((size_t)b * H_ + (n >> 6)) * NQ_ + m) * HD_ + (n & 63)] = (__bf16)v;
        } else if (mode == 1) {
          ((__bf16*)outp)[(((size_t)b * H_ + (n >> 6)) * CTX_ + m) * HD_ + (n & 63)] = (__bf16)v;
        } else if (mode == 2) {
          ((__bf16*)outp)[(((size_t)b * H_ + (n >> 6)) * HD_ + (n & 63)) * CTX_ + m] = (__bf16)v;
        } else {
          size_t o = ((size_t)b * C_ + n) * NQ_ + m;
          ((float*)outp)[o] = v + bo[n] + xres[o];
        }
      }
    }
  }
}

// ---------------- flash attention ----------------
// One wave per 32-query tile (each of the 32 lanes owns exactly one softmax
// row -> no EXEC divergence). K stepped by 32 context rows; K/V fragments are
// reused across both 16-row M-tiles: 16 WMMAs per step for 16 b128 K/V loads.
__global__ void __launch_bounds__(128)
attn_kernel(const __bf16* __restrict__ Q, const __bf16* __restrict__ Kb,
            const __bf16* __restrict__ Vt, __bf16* __restrict__ O) {
  __shared__ float Sbuf[4][32][33];
  __shared__ __align__(16) __bf16 Pbuf[4][32][32];
  __shared__ float rowscale[4][32];
  __shared__ float mrow[4][32];
  __shared__ float lrow[4][32];

  int wave = threadIdx.x >> 5;
  int lane = threadIdx.x & 31;
  int t = blockIdx.x * 4 + wave;      // 2048 tiles total
  int qt = t & 127;                   // 128 query tiles of 32 rows
  int h = (t >> 7) & 7;
  int b = t >> 10;

  const __bf16* Qp = Q + (((size_t)b * H_ + h) * NQ_ + (size_t)qt * 32) * HD_;
  const __bf16* Kp = Kb + ((size_t)b * H_ + h) * CTX_ * HD_;
  const __bf16* Vp = Vt + ((size_t)b * H_ + h) * HD_ * CTX_;

  // Q fragments: [m-tile][k-chunk]
  v16bf qa[2][2];
#pragma unroll
  for (int mi = 0; mi < 2; ++mi) {
    qa[mi][0] = load_a_frag(Qp + (size_t)mi * 16 * HD_ + 0, HD_);
    qa[mi][1] = load_a_frag(Qp + (size_t)mi * 16 * HD_ + 32, HD_);
  }

  v8f o[2][4];
#pragma unroll
  for (int mi = 0; mi < 2; ++mi)
#pragma unroll
    for (int i = 0; i < 4; ++i) o[mi][i] = zero8();

  int col = lane & 15;
  int rowoff = (lane >= 16) ? 8 : 0;

  mrow[wave][lane] = -3.0e38f;
  lrow[wave][lane] = 0.f;
  __syncthreads();

  for (int kb = 0; kb < CTX_; kb += 32) {
    __builtin_prefetch(Kp + (size_t)(kb + 32) * HD_, 0, 1);
    // S tiles (32q x 32k): K fragments shared across both M-tiles
    v8f s[2][2];
#pragma unroll
    for (int ni = 0; ni < 2; ++ni) {
      v16bf bk0 = load_b_frag(Kp + (size_t)(kb + ni * 16) * HD_ + 0, HD_);
      v16bf bk1 = load_b_frag(Kp + (size_t)(kb + ni * 16) * HD_ + 32, HD_);
#pragma unroll
      for (int mi = 0; mi < 2; ++mi) {
        v8f sv = zero8();
        sv = wmma_bf16(qa[mi][0], bk0, sv);
        sv = wmma_bf16(qa[mi][1], bk1, sv);
        s[mi][ni] = sv;
      }
    }
#pragma unroll
    for (int mi = 0; mi < 2; ++mi)
#pragma unroll
      for (int r = 0; r < 8; ++r) {
        Sbuf[wave][mi * 16 + r + rowoff][col] = s[mi][0][r];
        Sbuf[wave][mi * 16 + r + rowoff][16 + col] = s[mi][1][r];
      }
    __syncthreads();

    // online softmax: lane L owns row L (full wave, no divergence)
    {
      int row = lane;
      float m0 = mrow[wave][row];
      float mx = m0;
#pragma unroll 8
      for (int j = 0; j < 32; ++j) mx = fmaxf(mx, Sbuf[wave][row][j] * SCL_);
      float sc = __expf(m0 - mx);
      float sum = 0.f;
#pragma unroll 8
      for (int j = 0; j < 32; ++j) {
        float p = __expf(Sbuf[wave][row][j] * SCL_ - mx);
        Pbuf[wave][row][j] = (__bf16)p;
        sum += p;
      }
      mrow[wave][row] = mx;
      lrow[wave][row] = lrow[wave][row] * sc + sum;
      rowscale[wave][row] = sc;
    }
    __syncthreads();

    // rescale running output
#pragma unroll
    for (int mi = 0; mi < 2; ++mi)
#pragma unroll
      for (int r = 0; r < 8; ++r) {
        float sc = rowscale[wave][mi * 16 + r + rowoff];
        o[mi][0][r] *= sc; o[mi][1][r] *= sc; o[mi][2][r] *= sc; o[mi][3][r] *= sc;
      }

    // P as A-fragments from LDS (one per M-tile)
    v16bf pa[2];
#pragma unroll
    for (int mi = 0; mi < 2; ++mi) {
      int m = lane & 15;
      int kbp = (lane >= 16) ? 8 : 0;
      const __bf16* pp = &Pbuf[wave][mi * 16 + m][kbp];
      bf16x8 lo = *(const bf16x8*)(pp);
      bf16x8 hi = *(const bf16x8*)(pp + 16);
#pragma unroll
      for (int i = 0; i < 8; ++i) { pa[mi][i] = lo[i]; pa[mi][i + 8] = hi[i]; }
    }

    // O(32x64) += P(32x32) x V(32x64); V fragments shared across M-tiles
#pragma unroll
    for (int t4 = 0; t4 < 4; ++t4) {
      v16bf bv = load_b_frag(Vp + (size_t)(t4 * 16) * CTX_ + kb, CTX_);
      o[0][t4] = wmma_bf16(pa[0], bv, o[0][t4]);
      o[1][t4] = wmma_bf16(pa[1], bv, o[1][t4]);
    }
    __syncthreads();
  }

  rowscale[wave][lane] = 1.f / lrow[wave][lane];
  __syncthreads();

#pragma unroll
  for (int mi = 0; mi < 2; ++mi)
#pragma unroll
    for (int t4 = 0; t4 < 4; ++t4)
#pragma unroll
      for (int r = 0; r < 8; ++r) {
        int row = mi * 16 + r + rowoff;
        float v = o[mi][t4][r] * rowscale[wave][row];
        O[((size_t)b * NQ_ + (size_t)qt * 32 + row) * C_ + h * HD_ + t4 * 16 + col] =
            (__bf16)v;
      }
}

// ---------------- host launcher ----------------

extern "C" void kernel_launch(void* const* d_in, const int* in_sizes, int n_in,
                              void* d_out, int out_size, void* d_ws, size_t ws_size,
                              hipStream_t stream) {
  const float* x = (const float*)d_in[0];
  const float* ctx = (const float*)d_in[1];
  const float* gamma = (const float*)d_in[2];
  const float* beta = (const float*)d_in[3];
  const float* Wq = (const float*)d_in[4];
  const float* Wk = (const float*)d_in[5];
  const float* Wv = (const float*)d_in[6];
  const float* Wo = (const float*)d_in[7];
  const float* bo = (const float*)d_in[8];
  float* out = (float*)d_out;

  char* ws = (char*)d_ws;
  size_t off = 0;
  auto alloc = [&](size_t bytes) -> void* {
    void* p = ws + off;
    off = (off + bytes + 255) & ~(size_t)255;
    return p;
  };

  float* stats = (float*)alloc((size_t)B_ * G_ * 2 * sizeof(float));
  __bf16* xnT = (__bf16*)alloc((size_t)B_ * NQ_ * C_ * 2);
  __bf16* ctxb = (__bf16*)alloc((size_t)B_ * CTX_ * CD_ * 2);
  __bf16* WqT = (__bf16*)alloc((size_t)C_ * C_ * 2);
  __bf16* WkT = (__bf16*)alloc((size_t)C_ * CD_ * 2);
  __bf16* WvT = (__bf16*)alloc((size_t)C_ * CD_ * 2);
  __bf16* WoT = (__bf16*)alloc((size_t)C_ * C_ * 2);
  __bf16* Qb = (__bf16*)alloc((size_t)B_ * H_ * NQ_ * HD_ * 2);
  __bf16* Kbf = (__bf16*)alloc((size_t)B_ * H_ * CTX_ * HD_ * 2);
  __bf16* Vtb = (__bf16*)alloc((size_t)B_ * H_ * HD_ * CTX_ * 2);
  __bf16* Ob = (__bf16*)alloc((size_t)B_ * NQ_ * C_ * 2);

  // 1) GroupNorm
  gn_stats_kernel<<<B_ * G_, 256, 0, stream>>>(x, stats);
  {
    int total = B_ * NQ_ * C_;
    gn_apply_kernel<<<(total + 255) / 256, 256, 0, stream>>>(x, stats, gamma, beta, xnT);
  }

  // 2) format conversions
  {
    int n = B_ * CTX_ * CD_;
    convert_bf16_kernel<<<(n + 255) / 256, 256, 0, stream>>>(ctx, ctxb, n);
  }
  transpose_bf16_kernel<<<(C_ * C_ + 255) / 256, 256, 0, stream>>>(Wq, WqT, C_, C_);
  transpose_bf16_kernel<<<(CD_ * C_ + 255) / 256, 256, 0, stream>>>(Wk, WkT, CD_, C_);
  transpose_bf16_kernel<<<(CD_ * C_ + 255) / 256, 256, 0, stream>>>(Wv, WvT, CD_, C_);
  transpose_bf16_kernel<<<(C_ * C_ + 255) / 256, 256, 0, stream>>>(Wo, WoT, C_, C_);

  // 3) projections (waves = B * (M/32) * 8; 4 waves per block)
  gemm_wmma_kernel<<<(B_ * (NQ_ / 32) * 8) / 4, 128, 0, stream>>>(
      xnT, WqT, NQ_, C_, NQ_ * C_, 0, Qb, nullptr, nullptr);
  gemm_wmma_kernel<<<(B_ * (CTX_ / 32) * 8) / 4, 128, 0, stream>>>(
      ctxb, WkT, CTX_, CD_, CTX_ * CD_, 1, Kbf, nullptr, nullptr);
  gemm_wmma_kernel<<<(B_ * (CTX_ / 32) * 8) / 4, 128, 0, stream>>>(
      ctxb, WvT, CTX_, CD_, CTX_ * CD_, 2, Vtb, nullptr, nullptr);

  // 4) attention: B*H*(NQ/32) = 2048 waves, 4 per block
  attn_kernel<<<(B_ * H_ * (NQ_ / 32)) / 4, 128, 0, stream>>>(Qb, Kbf, Vtb, Ob);

  // 5) output projection + bias + residual, write [b, c, n] f32
  gemm_wmma_kernel<<<(B_ * (NQ_ / 32) * 8) / 4, 128, 0, stream>>>(
      Ob, WoT, NQ_, C_, NQ_ * C_, 3, out, bo, x);
}
